// CellTrack_Model_26233660244112
// MI455X (gfx1250) — matrix operand order repaired and checked by reading, not compiled, not run on hardware
//
#include <hip/hip_runtime.h>
#include <hip/hip_fp16.h>

// ---------------------------------------------------------------------------
// CellTrack GNN on MI455X (gfx1250), wave32 + v_wmma_f32_16x16x32_f16.
// GEMMs are computed operand-swapped: D = W^T x Act^T, so A = weights
// (K-major arena rows = output channels) and B = activations (row-per-edge,
// K-contiguous LDS tile). The D fragment then gives each lane 8 CONSECUTIVE
// output channels of one edge -> epilogue is packed cvt + one b128 store
// straight to global memory (coalesced), no LDS round trip.
// ---------------------------------------------------------------------------

typedef __attribute__((ext_vector_type(16))) _Float16 v16h;
typedef __attribute__((ext_vector_type(8)))  _Float16 v8h;
typedef __attribute__((ext_vector_type(8)))  float    v8f;
typedef __attribute__((ext_vector_type(4)))  float    v4f;

#define N_NODES 50000
#define N_EDGES 800000

// ---- WMMA fragment loaders (CDNA5 16-bit layouts, CDNA5 ISA) ----

// A-side = weights, K-major LDS Wl[Nch][ldk]; tile t covers channels
// [m0, m0+16). Lane r=L&15 is row M; h=L>>4: two contiguous 16B K-runs.
__device__ __forceinline__ v16h frag_w(const _Float16* Wl, int ldk, int lane,
                                       int k0, int m0) {
    const int r = lane & 15, h = lane >> 4;
    const _Float16* p = Wl + (size_t)(m0 + r) * ldk + k0 + (h << 3);
    const v8h lo = *(const v8h*)(p);
    const v8h hi = *(const v8h*)(p + 16);
    return __builtin_shufflevector(lo, hi, 0, 1, 2, 3, 4, 5, 6, 7,
                                   8, 9, 10, 11, 12, 13, 14, 15);
}

// B-side = activations, row-per-edge K-contiguous LDS Act[16][ldk]:
// lane c=L&15 is column N (edge); element j<8 -> K = k0 + h*8 + j, j>=8 -> K = k0 + 16 + h*8 + (j-8).
__device__ __forceinline__ v16h frag_act(const _Float16* Act, int ldk, int lane, int k0) {
    const int c = lane & 15, h = lane >> 4;
    const _Float16* p = Act + (size_t)c * ldk + k0 + (h << 3);
    const v8h lo = *(const v8h*)(p);
    const v8h hi = *(const v8h*)(p + 16);
    return __builtin_shufflevector(lo, hi, 0, 1, 2, 3, 4, 5, 6, 7,
                                   8, 9, 10, 11, 12, 13, 14, 15);
}

// D 16x16 f32: VGPR i -> channel M = m0 + i + 8*(lane>>4); edge N = lane&15.
// => per lane & tile: 8 consecutive channels of one edge.

// acc[NT] (channels NT*16) += W[ch x K] x Act[K x 16]
template <int NT>
__device__ __forceinline__ void gemm_wa(v8f* acc, const _Float16* Wl, int wldk, int wK0,
                                        const _Float16* Act, int aldk, int aK0,
                                        int ksteps, int lane) {
    for (int ks = 0; ks < ksteps; ++ks) {
        const v16h b = frag_act(Act, aldk, lane, aK0 + ks * 32);
#pragma unroll
        for (int t = 0; t < NT; ++t) {
            const v16h a = frag_w(Wl, wldk, lane, wK0 + ks * 32, t * 16);
            acc[t] = __builtin_amdgcn_wmma_f32_16x16x32_f16(
                false, a, false, b, (short)0, acc[t], false, false);
        }
    }
}

// bias(+relu) + pack 8 consecutive channels [ch0, ch0+8) to f16.
__device__ __forceinline__ v8h epilogue8(const v8f acc, const float* bl,
                                         int ch0, bool relu) {
    v8h o;
#pragma unroll
    for (int i = 0; i < 8; ++i) {
        float v = acc[i] + bl[ch0 + i];
        if (relu) v = fmaxf(v, 0.0f);
        o[i] = (_Float16)v;
    }
    return o;
}

// Pure vector copy global f16 -> LDS f16 (n_halves multiple of 8).
__device__ __forceinline__ void stage_cp(_Float16* dst, const _Float16* src,
                                         int n_halves, int tid, int nthr) {
    const int nv = n_halves >> 3;
    for (int i = tid; i < nv; i += nthr)
        ((v8h*)dst)[i] = ((const v8h*)src)[i];
}

// Copy one 64-half (128B) row with v8h chunks.
__device__ __forceinline__ void copy_row64(_Float16* dst, const _Float16* src) {
#pragma unroll
    for (int j = 0; j < 8; ++j) ((v8h*)dst)[j] = ((const v8h*)src)[j];
}

// Convert 8 f32 (two v4f) -> one v8h.
__device__ __forceinline__ v8h cvt8(const v4f a, const v4f b) {
    v8h h;
#pragma unroll
    for (int u = 0; u < 4; ++u) {
        h[u] = (_Float16)a[u];
        h[4 + u] = (_Float16)b[u];
    }
    return h;
}

// ---------------------------------------------------------------------------
// Prep kernels: f32 W[Krows x Nc] -> f16 K-major Wt[Nc][Kpad], zero-padded K.
// ---------------------------------------------------------------------------
__global__ void k_prep_wT(_Float16* __restrict__ dst, const float* __restrict__ W,
                          int Krows, int Kpad, int Nc) {
    const int i = blockIdx.x * blockDim.x + threadIdx.x;
    if (i < Nc * Kpad) {
        const int n = i / Kpad, k = i - n * Kpad;
        dst[i] = (k < Krows) ? (_Float16)W[k * Nc + n] : (_Float16)0.0f;
    }
}

// edge_enc W0 with the feature permutation folded in:
// LDS feature order: [0..63]=xs [64..127]=xt [128..191]=|xs-xt|
//                    [192..223]=abs_init [224]=sim1 [225]=sim2 [226..255]=pad
__device__ __forceinline__ int enc_row_map(int k) {
    if (k < 64)  return 33 + k;
    if (k < 128) return 97 + (k - 64);
    if (k < 192) return 161 + (k - 128);
    if (k < 224) return k - 192;
    if (k == 224) return 32;
    if (k == 225) return 225;
    return -1;
}

__global__ void k_prep_encW0(_Float16* __restrict__ dst, const float* __restrict__ W0) {
    const int i = blockIdx.x * blockDim.x + threadIdx.x;
    if (i < 128 * 256) {
        const int n = i >> 8, k = i & 255;
        const int m = enc_row_map(k);
        dst[i] = (m >= 0) ? (_Float16)W0[m * 128 + n] : (_Float16)0.0f;
    }
}

// ---------------------------------------------------------------------------
// Kernel 1: node MLPs  cf = MLP(hand, feat) [16->64->32], cp likewise.
// x[N,64] f16 = concat(cf, cp).  One 16-node tile per wave, 4 waves/block.
// ---------------------------------------------------------------------------
__global__ void __launch_bounds__(128) k_node_mlp(
    const float* __restrict__ cell_feat, const float* __restrict__ cell_param,
    const _Float16* hW0t, const float* hb0, const _Float16* hW1t, const float* hb1,
    const _Float16* lW0t, const float* lb0, const _Float16* lW1t, const float* lb1,
    _Float16* __restrict__ x) {
    __shared__ _Float16 W0l[64 * 32];   // [64 ch][32 k]
    __shared__ _Float16 W1l[32 * 64];   // [32 ch][64 k]
    __shared__ float    b0l[64], b1l[32];
    __shared__ _Float16 A[4][16 * 64];

    const int tid = threadIdx.x, nthr = blockDim.x;
    const int lane = tid & 31, wave = tid >> 5;
    const int node0 = (blockIdx.x * 4 + wave) * 16;
    const int c = lane & 15, h = lane >> 4;
    _Float16* Aw = A[wave];

    for (int pass = 0; pass < 2; ++pass) {
        const float* feat = pass ? cell_param : cell_feat;
        const _Float16* W0t = pass ? lW0t : hW0t;  const float* b0g = pass ? lb0 : hb0;
        const _Float16* W1t = pass ? lW1t : hW1t;  const float* b1g = pass ? lb1 : hb1;

        stage_cp(W0l, W0t, 64 * 32, tid, nthr);
        stage_cp(W1l, W1t, 32 * 64, tid, nthr);
        for (int n = tid; n < 64; n += nthr) b0l[n] = b0g[n];
        for (int n = tid; n < 32; n += nthr) b1l[n] = b1g[n];
        __syncthreads();

        // Act tile [16 nodes x 32k] (K=16 zero-padded to 32), ldk=64
        if (lane < 16) {
            const int node = node0 + lane;
            _Float16* row = Aw + lane * 64;
            const v8h z = {};
            if (node < N_NODES) {
                const v4f* f4 = (const v4f*)(feat + (size_t)node * 16);
                ((v8h*)row)[0] = cvt8(f4[0], f4[1]);
                ((v8h*)row)[1] = cvt8(f4[2], f4[3]);
            } else {
                ((v8h*)row)[0] = z; ((v8h*)row)[1] = z;
            }
            ((v8h*)row)[2] = z; ((v8h*)row)[3] = z;   // K pad 16..31
        }
        // layer 1: 64ch x K32 -> hid (relu) back into Aw rows, K-contiguous
        v8f acc[4] = {};
        gemm_wa<4>(acc, W0l, 32, 0, Aw, 64, 0, 1, lane);
#pragma unroll
        for (int t = 0; t < 4; ++t) {
            const int ch0 = t * 16 + h * 8;
            *(v8h*)(Aw + c * 64 + ch0) = epilogue8(acc[t], b0l, ch0, true);
        }
        // layer 2: 32ch x K64 -> cf/cp (no relu), direct global store
        v8f acc2[2] = {};
        gemm_wa<2>(acc2, W1l, 64, 0, Aw, 64, 0, 2, lane);
        const int node = node0 + c;
#pragma unroll
        for (int t = 0; t < 2; ++t) {
            const int ch0 = t * 16 + h * 8;
            const v8h o = epilogue8(acc2[t], b1l, ch0, false);
            if (node < N_NODES)
                *(v8h*)(x + (size_t)node * 64 + pass * 32 + ch0) = o;
        }
        __syncthreads();  // before restaging weights for next pass
    }
}

// ---------------------------------------------------------------------------
// Kernel 2: fused edge-feature construction + edge_enc MLP (226->128->64).
// ---------------------------------------------------------------------------
__global__ void __launch_bounds__(128) k_edge_enc(
    const float* __restrict__ cell_feat, const float* __restrict__ cell_param,
    const int* __restrict__ eidx, const _Float16* __restrict__ x,
    const _Float16* W0t,  // [128 ch][256 k] permuted
    const float* b0, const _Float16* W1t,  // [64 ch][128 k]
    const float* b1, _Float16* __restrict__ e) {
    __shared__ _Float16 F[4][16 * 256];
    __shared__ _Float16 Wl[128 * 64];    // K-chunked [Nch][64]
    __shared__ float    b0l[128], b1l[64];

    const int tid = threadIdx.x, nthr = blockDim.x;
    const int lane = tid & 31, wave = tid >> 5;
    const int e0 = (blockIdx.x * 4 + wave) * 16;
    const int c = lane & 15, h = lane >> 4;
    const int* srcv = eidx;
    const int* trgv = eidx + N_EDGES;
    _Float16* Fw = F[wave];

    for (int n = tid; n < 128; n += nthr) b0l[n] = b0[n];
    for (int n = tid; n < 64; n += nthr) b1l[n] = b1[n];

    if (lane < 16) {  // one edge per lane
        const int ed = e0 + lane;
        const int s = srcv[ed], t = trgv[ed];
        _Float16* row = Fw + lane * 256;
        const _Float16* xsp = x + (size_t)s * 64;
        const _Float16* xtp = x + (size_t)t * 64;
        float dot = 0.f, na = 0.f, nb = 0.f;
#pragma unroll
        for (int j = 0; j < 8; ++j) {
            const v8h a8 = ((const v8h*)xsp)[j];
            const v8h b8 = ((const v8h*)xtp)[j];
            ((v8h*)row)[j] = a8;
            ((v8h*)(row + 64))[j] = b8;
            v8h d8;
#pragma unroll
            for (int u = 0; u < 8; ++u) {
                const float xa = (float)a8[u], xb = (float)b8[u];
                d8[u] = (_Float16)fabsf(xa - xb);
                dot += xa * xb; na += xa * xa; nb += xb * xb;
            }
            ((v8h*)(row + 128))[j] = d8;
        }
        row[225] = (_Float16)(dot / fmaxf(sqrtf(na) * sqrtf(nb), 1e-8f));
        dot = 0.f; na = 0.f; nb = 0.f;
        for (int k = 0; k < 32; ++k) {
            const float a = (k < 16) ? cell_feat[s * 16 + k] : cell_param[s * 16 + (k - 16)];
            const float b = (k < 16) ? cell_feat[t * 16 + k] : cell_param[t * 16 + (k - 16)];
            row[192 + k] = (_Float16)fabsf(a - b);
            dot += a * b; na += a * a; nb += b * b;
        }
        row[224] = (_Float16)(dot / fmaxf(sqrtf(na) * sqrtf(nb), 1e-8f));
        for (int k = 226; k < 256; ++k) row[k] = (_Float16)0.0f;
    }
    __syncthreads();

    // layer 1: 128ch x K256, K chunked 4x64 (b128 copy staging)
    v8f acc[8] = {};
    for (int ck = 0; ck < 4; ++ck) {
        for (int i = tid; i < 128 * 8; i += nthr) {
            const int n = i >> 3, j = i & 7;
            ((v8h*)Wl)[i] = ((const v8h*)(W0t + n * 256 + ck * 64))[j];
        }
        __syncthreads();
        gemm_wa<8>(acc, Wl, 64, 0, Fw, 256, ck * 64, 2, lane);
        __syncthreads();
    }
    // hid = relu(acc + b0) -> Fw rows, K-contiguous: one b128 store per tile
#pragma unroll
    for (int t = 0; t < 8; ++t) {
        const int ch0 = t * 16 + h * 8;
        *(v8h*)(Fw + c * 256 + ch0) = epilogue8(acc[t], b0l, ch0, true);
    }

    // layer 2: 64ch x K128, K chunked 2x64, no relu, direct global store
    v8f acc2[4] = {};
    for (int ck = 0; ck < 2; ++ck) {
        for (int i = tid; i < 64 * 8; i += nthr) {
            const int n = i >> 3, j = i & 7;
            ((v8h*)Wl)[i] = ((const v8h*)(W1t + n * 128 + ck * 64))[j];
        }
        __syncthreads();
        gemm_wa<4>(acc2, Wl, 64, 0, Fw, 256, ck * 64, 2, lane);
        __syncthreads();
    }
#pragma unroll
    for (int t = 0; t < 4; ++t) {
        const int ch0 = t * 16 + h * 8;
        *(v8h*)(e + (size_t)(e0 + c) * 64 + ch0) = epilogue8(acc2[t], b1l, ch0, false);
    }
}

// ---------------------------------------------------------------------------
// Kernel 3: zero agg
// ---------------------------------------------------------------------------
__global__ void k_zero(float* __restrict__ p, int n) {
    const int i = blockIdx.x * blockDim.x + threadIdx.x;
    if (i < n) p[i] = 0.0f;
}

// ---------------------------------------------------------------------------
// Kernel 4: msg MLP (192->64) + fused segment_sum scatter (atomicAdd f32).
// Each lane adds 8 consecutive agg channels of one edge's target node.
// ---------------------------------------------------------------------------
__global__ void __launch_bounds__(128) k_msg(
    const int* __restrict__ eidx, const _Float16* __restrict__ x,
    const _Float16* __restrict__ e, const _Float16* Wt, const float* b,
    float* __restrict__ agg) {
    __shared__ _Float16 A[4][16 * 192];
    __shared__ _Float16 Wl[64 * 192];
    __shared__ float    bl[64];

    const int tid = threadIdx.x, nthr = blockDim.x;
    const int lane = tid & 31, wave = tid >> 5;
    const int e0 = (blockIdx.x * 4 + wave) * 16;
    const int c = lane & 15, h = lane >> 4;
    const int* srcv = eidx;
    const int* trgv = eidx + N_EDGES;
    _Float16* Aw = A[wave];

    stage_cp(Wl, Wt, 64 * 192, tid, nthr);
    for (int n = tid; n < 64; n += nthr) bl[n] = b[n];

    if (lane < 16) {
        const int ed = e0 + lane;
        const int s = srcv[ed], t = trgv[ed];
        _Float16* row = Aw + lane * 192;
        copy_row64(row,       x + (size_t)s * 64);
        copy_row64(row + 64,  x + (size_t)t * 64);
        copy_row64(row + 128, e + (size_t)ed * 64);
    }
    __syncthreads();

    v8f acc[4] = {};
    gemm_wa<4>(acc, Wl, 192, 0, Aw, 192, 0, 6, lane);

    const int tg = trgv[e0 + c];
    float* aggrow = agg + (size_t)tg * 64;
#pragma unroll
    for (int t = 0; t < 4; ++t) {
        const int ch0 = t * 16 + h * 8;
#pragma unroll
        for (int i = 0; i < 8; ++i)
            atomicAdd(aggrow + ch0 + i, fmaxf(acc[t][i] + bl[ch0 + i], 0.0f));
    }
}

// ---------------------------------------------------------------------------
// Kernel 5: node update MLP (128->64), relu, in-place x, direct stores.
// ---------------------------------------------------------------------------
__global__ void __launch_bounds__(128) k_node_upd(
    _Float16* __restrict__ x, const float* __restrict__ agg,
    const _Float16* Wt, const float* b) {
    __shared__ _Float16 A[4][16 * 128];
    __shared__ _Float16 Wl[64 * 128];
    __shared__ float    bl[64];

    const int tid = threadIdx.x, nthr = blockDim.x;
    const int lane = tid & 31, wave = tid >> 5;
    const int node0 = (blockIdx.x * 4 + wave) * 16;
    const int c = lane & 15, h = lane >> 4;
    _Float16* Aw = A[wave];

    stage_cp(Wl, Wt, 64 * 128, tid, nthr);
    for (int n = tid; n < 64; n += nthr) bl[n] = b[n];

    if (lane < 16) {
        const int node = node0 + lane;
        _Float16* row = Aw + lane * 128;
        if (node < N_NODES) {
            copy_row64(row, x + (size_t)node * 64);
            const v4f* a4 = (const v4f*)(agg + (size_t)node * 64);
#pragma unroll
            for (int j = 0; j < 8; ++j)
                ((v8h*)(row + 64))[j] = cvt8(a4[2 * j], a4[2 * j + 1]);
        } else {
            const v8h z = {};
#pragma unroll
            for (int j = 0; j < 16; ++j) ((v8h*)row)[j] = z;
        }
    }
    __syncthreads();

    v8f acc[4] = {};
    gemm_wa<4>(acc, Wl, 128, 0, Aw, 128, 0, 4, lane);

    const int node = node0 + c;
#pragma unroll
    for (int t = 0; t < 4; ++t) {
        const int ch0 = t * 16 + h * 8;
        const v8h o = epilogue8(acc[t], bl, ch0, true);
        if (node < N_NODES)
            *(v8h*)(x + (size_t)node * 64 + ch0) = o;
    }
}

// ---------------------------------------------------------------------------
// Kernel 6: edge update MLP (192->64), relu, in-place e, direct stores.
// ---------------------------------------------------------------------------
__global__ void __launch_bounds__(128) k_edge_upd(
    const int* __restrict__ eidx, const _Float16* __restrict__ x,
    _Float16* __restrict__ e, const _Float16* Wt, const float* b) {
    __shared__ _Float16 A[4][16 * 192];
    __shared__ _Float16 Wl[64 * 192];
    __shared__ float    bl[64];

    const int tid = threadIdx.x, nthr = blockDim.x;
    const int lane = tid & 31, wave = tid >> 5;
    const int e0 = (blockIdx.x * 4 + wave) * 16;
    const int c = lane & 15, h = lane >> 4;
    const int* srcv = eidx;
    const int* trgv = eidx + N_EDGES;
    _Float16* Aw = A[wave];

    stage_cp(Wl, Wt, 64 * 192, tid, nthr);
    for (int n = tid; n < 64; n += nthr) bl[n] = b[n];

    if (lane < 16) {
        const int ed = e0 + lane;
        const int s = srcv[ed], t = trgv[ed];
        _Float16* row = Aw + lane * 192;
        copy_row64(row,       e + (size_t)ed * 64);
        copy_row64(row + 64,  x + (size_t)s * 64);
        copy_row64(row + 128, x + (size_t)t * 64);
    }
    __syncthreads();

    v8f acc[4] = {};
    gemm_wa<4>(acc, Wl, 192, 0, Aw, 192, 0, 6, lane);
#pragma unroll
    for (int t = 0; t < 4; ++t) {
        const int ch0 = t * 16 + h * 8;
        *(v8h*)(e + (size_t)(e0 + c) * 64 + ch0) = epilogue8(acc[t], bl, ch0, true);
    }
}

// ---------------------------------------------------------------------------
// Kernel 7: classifier (64->64 relu, 64->1) -> pred f32.
// hid stays in registers; cross-half reduce via shfl_xor(16).
// ---------------------------------------------------------------------------
__global__ void __launch_bounds__(128) k_clf(
    const _Float16* __restrict__ e, const _Float16* W0t, const float* b0,
    const float* W1, const float* b1, float* __restrict__ out) {
    __shared__ _Float16 A[4][16 * 64];
    __shared__ _Float16 Wl[64 * 64];
    __shared__ float    b0l[64], W1l[64], b1l;

    const int tid = threadIdx.x, nthr = blockDim.x;
    const int lane = tid & 31, wave = tid >> 5;
    const int e0 = (blockIdx.x * 4 + wave) * 16;
    const int c = lane & 15, h = lane >> 4;
    _Float16* Aw = A[wave];

    stage_cp(Wl, W0t, 64 * 64, tid, nthr);
    for (int n = tid; n < 64; n += nthr) { b0l[n] = b0[n]; W1l[n] = W1[n]; }
    if (tid == 0) b1l = b1[0];

#pragma unroll
    for (int it = 0; it < 4; ++it) {     // 16x64 tile is contiguous: 128 v8h
        const int id = lane + 32 * it;
        ((v8h*)Aw)[id] = ((const v8h*)(e + (size_t)e0 * 64))[id];
    }
    __syncthreads();

    v8f acc[4] = {};
    gemm_wa<4>(acc, Wl, 64, 0, Aw, 64, 0, 2, lane);

    // pred = b1 + sum_ch relu(hid)*W1; lane holds 32 of 64 channels of edge c.
    float p = 0.0f;
#pragma unroll
    for (int t = 0; t < 4; ++t) {
        const int ch0 = t * 16 + h * 8;
#pragma unroll
        for (int i = 0; i < 8; ++i)
            p += fmaxf(acc[t][i] + b0l[ch0 + i], 0.0f) * W1l[ch0 + i];
    }
    p += __shfl_xor(p, 16, 32);          // combine halves h=0 and h=1
    if (lane < 16) out[e0 + c] = p + b1l;
}

// ---------------------------------------------------------------------------
// Host launch. Param flattening order (setup_inputs dict order):
//  0 cell_feat  1 cell_param  2 edge_index
//  3..6   hand   W0[16x64] b0[64] W1[64x32] b1[32]
//  7..10  learned (same shapes)
//  11..14 edge_enc W0[226x128] b0[128] W1[128x64] b1[64]
//  15+6L..20+6L  gnn[L]: msgW[192x64] msgb nodeW[128x64] nodeb edgeW[192x64] edgeb
//  33..36 clf W0[64x64] b0[64] W1[64x1] b1[1]
// ---------------------------------------------------------------------------
extern "C" void kernel_launch(void* const* d_in, const int* in_sizes, int n_in,
                              void* d_out, int out_size, void* d_ws, size_t ws_size,
                              hipStream_t stream) {
    (void)in_sizes; (void)out_size; (void)ws_size;
    auto fp = [&](int i) -> const float* {
        return (const float*)d_in[i < n_in ? i : (n_in - 1)];
    };
    const float* cell_feat  = fp(0);
    const float* cell_param = fp(1);
    const int*   eidx       = (const int*)d_in[2 < n_in ? 2 : 0];

    // Workspace layout (256B aligned):
    //   x f16 [N,64] | e f16 [E,64] | agg f32 [N,64] | f16 weight arena (~303KB)
    char* ws = (char*)d_ws;
    _Float16* x   = (_Float16*)ws;
    _Float16* e   = (_Float16*)(ws + (size_t)N_NODES * 64 * 2);
    float*    agg = (float*)(ws + (size_t)N_NODES * 64 * 2 + (size_t)N_EDGES * 64 * 2);
    _Float16* wb  = (_Float16*)(ws + (size_t)N_NODES * 64 * 2 + (size_t)N_EDGES * 64 * 2
                                   + (size_t)N_NODES * 64 * 4);

    // Weight arena offsets (halves)
    _Float16* hW0t = wb + 0;        // [64][32]
    _Float16* hW1t = wb + 2048;     // [32][64]
    _Float16* lW0t = wb + 4096;
    _Float16* lW1t = wb + 6144;
    _Float16* eW0t = wb + 8192;     // [128][256] permuted
    _Float16* eW1t = wb + 40960;    // [64][128]
    _Float16* cW0t = wb + 147456;   // [64][64]

    auto prep = [&](_Float16* dst, const float* W, int Krows, int Kpad, int Nc) {
        const int n = Nc * Kpad;
        k_prep_wT<<<(n + 255) / 256, 256, 0, stream>>>(dst, W, Krows, Kpad, Nc);
    };
    prep(hW0t, fp(3), 16, 32, 64);
    prep(hW1t, fp(5), 64, 64, 32);
    prep(lW0t, fp(7), 16, 32, 64);
    prep(lW1t, fp(9), 64, 64, 32);
    k_prep_encW0<<<(128 * 256 + 255) / 256, 256, 0, stream>>>(eW0t, fp(11));
    prep(eW1t, fp(13), 128, 128, 64);
    _Float16* gWt[3][3];
    for (int L = 0; L < 3; ++L) {
        _Float16* base = wb + 49152 + (size_t)L * 32768;
        gWt[L][0] = base;           // msg  [64][192]
        gWt[L][1] = base + 12288;   // node [64][128]
        gWt[L][2] = base + 20480;   // edge [64][192]
        const int pb = 15 + 6 * L;
        prep(gWt[L][0], fp(pb),     192, 192, 64);
        prep(gWt[L][1], fp(pb + 2), 128, 128, 64);
        prep(gWt[L][2], fp(pb + 4), 192, 192, 64);
    }
    prep(cW0t, fp(33), 64, 64, 64);

    const dim3 blk(128);
    const int node_blocks = ((N_NODES + 15) / 16 + 3) / 4;  // 782
    const int edge_blocks = (N_EDGES / 16) / 4;             // 12500

    k_node_mlp<<<node_blocks, blk, 0, stream>>>(
        cell_feat, cell_param,
        hW0t, fp(4), hW1t, fp(6),
        lW0t, fp(8), lW1t, fp(10), x);

    k_edge_enc<<<edge_blocks, blk, 0, stream>>>(
        cell_feat, cell_param, eidx, x,
        eW0t, fp(12), eW1t, fp(14), e);

    for (int L = 0; L < 3; ++L) {
        const int pb = 15 + 6 * L;
        k_zero<<<(N_NODES * 64 + 255) / 256, 256, 0, stream>>>(agg, N_NODES * 64);
        k_msg<<<edge_blocks, blk, 0, stream>>>(eidx, x, e, gWt[L][0], fp(pb + 1), agg);
        k_node_upd<<<node_blocks, blk, 0, stream>>>(x, agg, gWt[L][1], fp(pb + 3));
        k_edge_upd<<<edge_blocks, blk, 0, stream>>>(eidx, x, e, gWt[L][2], fp(pb + 5));
    }

    k_clf<<<edge_blocks, blk, 0, stream>>>(e, cW0t, fp(34), fp(35), fp(36),
                                           (float*)d_out);
}
